// HardMixingLoss_5136780886763
// MI455X (gfx1250) — matrix operand order repaired and checked by reading, not compiled, not run on hardware
//
#include <hip/hip_runtime.h>
#include <math.h>

typedef _Float16 v16h __attribute__((ext_vector_type(16)));
typedef _Float16 v8h  __attribute__((ext_vector_type(8)));
typedef float    v8f  __attribute__((ext_vector_type(8)));

#define NN   2048
#define DD   256
#define NP   4096
#define SS   150
#define THRK 204
#define THRP 208   // padded row stride for top-index table

__device__ inline v16h hcat16(v8h lo, v8h hi) {
    return __builtin_shufflevector(lo, hi, 0,1,2,3,4,5,6,7,8,9,10,11,12,13,14,15);
}
__device__ inline v8f wmma_f16(v16h a, v16h b, v8f c) {
    return __builtin_amdgcn_wmma_f32_16x16x32_f16(false, a, false, b, (short)0, c, false, false);
}

// ---------------------------------------------------------------------------
// Kernel 1: normalize pool rows -> f32 + split f16 (hi/lo) row-major
// ---------------------------------------------------------------------------
__global__ __launch_bounds__(256) void k_prep(const float* __restrict__ z1,
                                              const float* __restrict__ z2,
                                              float* __restrict__ znp,
                                              _Float16* __restrict__ zph,
                                              _Float16* __restrict__ zpl) {
    int r = blockIdx.x, tid = threadIdx.x;
    const float* src = (r < NN) ? z1 + (size_t)r * DD : z2 + (size_t)(r - NN) * DD;
    float v = src[tid];
    __shared__ float red[256];
    red[tid] = v * v;
    __syncthreads();
    for (int s = 128; s > 0; s >>= 1) { if (tid < s) red[tid] += red[tid + s]; __syncthreads(); }
    float inv = 1.0f / fmaxf(sqrtf(red[0]), 1e-12f);
    float zn = v * inv;
    znp[(size_t)r * DD + tid] = zn;
    _Float16 h = (_Float16)zn;
    zph[(size_t)r * DD + tid] = h;
    zpl[(size_t)r * DD + tid] = (_Float16)(zn - (float)h);
}

// ---------------------------------------------------------------------------
// Kernel 2: pack W1/W2 (D x D f32, row = k, col = n) into WMMA B-fragment
// layout: Wp[((kt*16+nt)*32 + lane)*16 + h], K=kt*32+(lane<16?0:8)+(h<8?h:h+8).
// ---------------------------------------------------------------------------
__global__ __launch_bounds__(256) void k_packW(const float* __restrict__ W1,
                                               const float* __restrict__ W2,
                                               _Float16* __restrict__ W1p,
                                               _Float16* __restrict__ W2p) {
    int t = blockIdx.x * 256 + threadIdx.x;          // 0 .. 2*65536-1
    int w = t >> 16, rem = t & 65535;
    int kt = rem >> 13, r2 = rem & 8191;
    int nt = r2 >> 9,  r3 = r2 & 511;
    int lane = r3 >> 4, h = r3 & 15;
    int k   = kt * 32 + ((lane >= 16) ? 8 : 0) + ((h < 8) ? h : h + 8);
    int col = nt * 16 + (lane & 15);
    const float* W = w ? W2 : W1;
    _Float16* Wp   = w ? W2p : W1p;
    Wp[rem] = (_Float16)W[k * DD + col];
}

// ---------------------------------------------------------------------------
// Kernel 3: Spool = Znorm @ Znorm^T (4096x4096), split-f16 WMMA for ~f32 acc.
// One 16x16 tile per wave; 8 waves/block.
// ---------------------------------------------------------------------------
__global__ __launch_bounds__(256) void k_sim(const _Float16* __restrict__ zph,
                                             const _Float16* __restrict__ zpl,
                                             float* __restrict__ Spool) {
    int tid = threadIdx.x, lane = tid & 31, wave = tid >> 5;
    int t = blockIdx.x * 8 + wave;
    int mt = t >> 8, nt = t & 255;
    int ln15 = lane & 15;
    bool hiL = lane >= 16;
    const _Float16* aph = zph + (size_t)(mt * 16 + ln15) * DD;
    const _Float16* apl = zpl + (size_t)(mt * 16 + ln15) * DD;
    const _Float16* bph = zph + (size_t)(nt * 16 + ln15) * DD;
    const _Float16* bpl = zpl + (size_t)(nt * 16 + ln15) * DD;
    v8f acc = {0.f,0.f,0.f,0.f,0.f,0.f,0.f,0.f};
    for (int kt = 0; kt < 8; ++kt) {
        int ka = kt * 32 + (hiL ? 8 : 0);
        int kb = kt * 32 + (hiL ? 8 : 0);
        v16h ah = hcat16(*(const v8h*)(aph + ka), *(const v8h*)(aph + ka + 16));
        v16h al = hcat16(*(const v8h*)(apl + ka), *(const v8h*)(apl + ka + 16));
        v16h bh = hcat16(*(const v8h*)(bph + kb), *(const v8h*)(bph + kb + 16));
        v16h bl = hcat16(*(const v8h*)(bpl + kb), *(const v8h*)(bpl + kb + 16));
        acc = wmma_f16(ah, bh, acc);    // hi*hi
        acc = wmma_f16(ah, bl, acc);    // hi*lo
        acc = wmma_f16(al, bh, acc);    // lo*hi  (lo*lo ~ 2^-22, dropped)
    }
    int col = nt * 16 + ln15;
    int rb  = mt * 16 + (hiL ? 8 : 0);
    for (int i = 0; i < 8; ++i)
        Spool[(size_t)(rb + i) * NP + col] = acc[i];
}

// ---------------------------------------------------------------------------
// Kernel 4: per-row sum(exp(2x)) + bitonic full sort of 4096 keys
// key = (~orderable(f32) << 32) | col  -> ascending sort = desc value, asc idx
// ---------------------------------------------------------------------------
__global__ __launch_bounds__(256) void k_sort(const float* __restrict__ Spool,
                                              float* __restrict__ sumexp,
                                              int* __restrict__ topIdx) {
    __shared__ unsigned long long sk[NP];   // 32 KB
    __shared__ float red[256];
    int r = blockIdx.x, tid = threadIdx.x;
    const float* row = Spool + (size_t)r * NP;
    float se = 0.f;
    for (int i = tid; i < NP; i += 256) {
        float x = row[i];
        se += expf(2.0f * x);               // f(x) = exp(x / 0.5)
        unsigned u = __float_as_uint(x);
        u ^= (u & 0x80000000u) ? 0xFFFFFFFFu : 0x80000000u;   // ascending-orderable
        unsigned key = ~u;                                     // descending
        sk[i] = ((unsigned long long)key << 32) | (unsigned)i;
    }
    red[tid] = se;
    __syncthreads();
    for (int s = 128; s > 0; s >>= 1) { if (tid < s) red[tid] += red[tid + s]; __syncthreads(); }
    if (tid == 0) sumexp[r] = red[0];
    for (unsigned k = 2; k <= NP; k <<= 1) {
        for (unsigned j = k >> 1; j > 0; j >>= 1) {
            for (unsigned i = tid; i < NP; i += 256) {
                unsigned ixj = i ^ j;
                if (ixj > i) {
                    unsigned long long a = sk[i], b = sk[ixj];
                    bool up = ((i & k) == 0);
                    if ((a > b) == up) { sk[i] = b; sk[ixj] = a; }
                }
            }
            __syncthreads();
        }
    }
    if (tid < THRK) topIdx[(size_t)r * THRP + tid] = (int)(sk[tid] & 0xFFFFFFFFull);
}

// ---------------------------------------------------------------------------
// Kernel 5: fused gather -> mixup -> FC1+ReLU -> FC2 -> cosine vs z -> sum exp
// One block per (view,row) r in [0,4096). 160x256 tile (150 real rows).
// ---------------------------------------------------------------------------
__global__ __launch_bounds__(256) void k_mixproj(const float* __restrict__ z1,
                                                 const float* __restrict__ z2,
                                                 const float* __restrict__ b1,
                                                 const float* __restrict__ b2,
                                                 const _Float16* __restrict__ W1p,
                                                 const _Float16* __restrict__ W2p,
                                                 const float* __restrict__ znp,
                                                 const int* __restrict__ idx1,
                                                 const int* __restrict__ idx2,
                                                 const int* __restrict__ topIdx,
                                                 float* __restrict__ negm) {
    __shared__ alignas(16) _Float16 sMix[160 * 256];   // 80 KB
    __shared__ alignas(16) _Float16 sH[160 * 256];     // 80 KB
    __shared__ float sDot[160], sSq[160];
    __shared__ int sG[2 * SS];
    __shared__ float sRed[256];

    int r = blockIdx.x, tid = threadIdx.x;
    int v = r >> 11, n = r & 2047;
    const int* idxv = v ? idx2 : idx1;
    const int* top  = topIdx + (size_t)r * THRP;

    if (tid < SS) {
        sG[tid]      = top[idxv[n * 2 * SS + tid]];
        sG[SS + tid] = top[idxv[n * 2 * SS + SS + tid]];
    }
    if (tid < 160) { sDot[tid] = 0.f; sSq[tid] = 0.f; }
    __syncthreads();

    // build mix tile: mix[s] = 0.2*z_pool[g1] + 0.8*z_pool[g2]; thread = column d
    for (int s = 0; s < SS; ++s) {
        int g1 = sG[s], g2 = sG[SS + s];
        const float* p1 = (g1 < NN) ? z1 + (size_t)g1 * DD : z2 + (size_t)(g1 - NN) * DD;
        const float* p2 = (g2 < NN) ? z1 + (size_t)g2 * DD : z2 + (size_t)(g2 - NN) * DD;
        sMix[s * 256 + tid] = (_Float16)(0.2f * p1[tid] + 0.8f * p2[tid]);
    }
    for (int s = SS; s < 160; ++s) sMix[s * 256 + tid] = (_Float16)0.f;
    __syncthreads();

    int lane = tid & 31, wave = tid >> 5;
    int ln15 = lane & 15;
    bool hiL = lane >= 16;

    // ---- layer 1: H = relu(mix @ W1 + b1) ----
    for (int t = wave; t < 160; t += 8) {
        int mt = t >> 4, nt = t & 15;
        v8f acc = {0.f,0.f,0.f,0.f,0.f,0.f,0.f,0.f};
        const _Float16* arow = sMix + (mt * 16 + ln15) * 256;
        for (int kt = 0; kt < 8; ++kt) {
            int ka = kt * 32 + (hiL ? 8 : 0);
            v16h a = hcat16(*(const v8h*)(arow + ka), *(const v8h*)(arow + ka + 16));
            const _Float16* bp = W1p + ((kt * 16 + nt) * 32 + lane) * 16;
            v16h b = hcat16(*(const v8h*)bp, *(const v8h*)(bp + 8));
            acc = wmma_f16(a, b, acc);
        }
        int col = nt * 16 + ln15;
        float bias = b1[col];
        int rb = mt * 16 + (hiL ? 8 : 0);
        for (int i = 0; i < 8; ++i) {
            float val = acc[i] + bias;
            val = val > 0.f ? val : 0.f;
            sH[(rb + i) * 256 + col] = (_Float16)val;
        }
    }
    __syncthreads();

    // ---- layer 2: h = H @ W2 + b2; epilogue: dot(h, zn[r]) and ||h||^2 ----
    const float* znr = znp + (size_t)r * DD;
    for (int t = wave; t < 160; t += 8) {
        int mt = t >> 4, nt = t & 15;
        v8f acc = {0.f,0.f,0.f,0.f,0.f,0.f,0.f,0.f};
        const _Float16* arow = sH + (mt * 16 + ln15) * 256;
        for (int kt = 0; kt < 8; ++kt) {
            int ka = kt * 32 + (hiL ? 8 : 0);
            v16h a = hcat16(*(const v8h*)(arow + ka), *(const v8h*)(arow + ka + 16));
            const _Float16* bp = W2p + ((kt * 16 + nt) * 32 + lane) * 16;
            v16h b = hcat16(*(const v8h*)bp, *(const v8h*)(bp + 8));
            acc = wmma_f16(a, b, acc);
        }
        int col = nt * 16 + ln15;
        float bias = b2[col];
        float znc = znr[col];
        int rb = mt * 16 + (hiL ? 8 : 0);
        for (int i = 0; i < 8; ++i) {
            float val = acc[i] + bias;
            float dp = val * znc;
            float sq = val * val;
            for (int m = 1; m < 16; m <<= 1) {   // reduce 16 cols of this tile
                dp += __shfl_xor(dp, m, 32);
                sq += __shfl_xor(sq, m, 32);
            }
            if (ln15 == 0) {
                atomicAdd(&sDot[rb + i], dp);
                atomicAdd(&sSq[rb + i], sq);
            }
        }
    }
    __syncthreads();

    float e = 0.f;
    if (tid < SS) {
        float ts = sDot[tid] / fmaxf(sqrtf(sSq[tid]), 1e-12f);
        e = expf(2.0f * ts);
    }
    sRed[tid] = e;
    __syncthreads();
    for (int s = 128; s > 0; s >>= 1) { if (tid < s) sRed[tid] += sRed[tid + s]; __syncthreads(); }
    if (tid == 0) negm[r] = sRed[0];
}

// ---------------------------------------------------------------------------
// Kernel 6: loss = mean_n 0.5*( log(d1)-2*pos + log(d2)-2*pos )
// ---------------------------------------------------------------------------
__global__ __launch_bounds__(256) void k_loss(const float* __restrict__ Spool,
                                              const float* __restrict__ sumexp,
                                              const float* __restrict__ negm,
                                              float* __restrict__ out) {
    __shared__ float red[256];
    int tid = threadIdx.x;
    float acc = 0.f;
    for (int n = tid; n < NN; n += 256) {
        float pos_s = Spool[(size_t)n * NP + NN + n];          // diag sim(z1,z2)
        float refl1 = Spool[(size_t)n * NP + n];
        float refl2 = Spool[(size_t)(NN + n) * NP + NN + n];
        float d1 = sumexp[n]      + negm[n]      - refl1;
        float d2 = sumexp[NN + n] + negm[NN + n] - refl2;
        acc += 0.5f * ((logf(d1) - 2.0f * pos_s) + (logf(d2) - 2.0f * pos_s));
    }
    red[tid] = acc;
    __syncthreads();
    for (int s = 128; s > 0; s >>= 1) { if (tid < s) red[tid] += red[tid + s]; __syncthreads(); }
    if (tid == 0) out[0] = red[0] / (float)NN;
}

// ---------------------------------------------------------------------------
extern "C" void kernel_launch(void* const* d_in, const int* in_sizes, int n_in,
                              void* d_out, int out_size, void* d_ws, size_t ws_size,
                              hipStream_t stream) {
    (void)in_sizes; (void)n_in; (void)out_size; (void)ws_size;
    const float* z1 = (const float*)d_in[0];
    const float* z2 = (const float*)d_in[1];
    const float* W1 = (const float*)d_in[2];
    const float* b1 = (const float*)d_in[3];
    const float* W2 = (const float*)d_in[4];
    const float* b2 = (const float*)d_in[5];
    const int* idx1 = (const int*)d_in[6];
    const int* idx2 = (const int*)d_in[7];

    char* ws = (char*)d_ws;
    float*    znp    = (float*)   (ws + 0);           //  4 MB  normalized pool f32
    _Float16* zph    = (_Float16*)(ws + 4194304);     //  2 MB  f16 hi
    _Float16* zpl    = (_Float16*)(ws + 6291456);     //  2 MB  f16 lo
    _Float16* W1p    = (_Float16*)(ws + 8388608);     //  128 KB packed W1
    _Float16* W2p    = (_Float16*)(ws + 8519680);     //  128 KB packed W2
    float*    Spool  = (float*)   (ws + 8650752);     //  64 MB 4096x4096 sims
    float*    sumexp = (float*)   (ws + 75759616);    //  16 KB
    int*      topIdx = (int*)     (ws + 75776000);    //  3.25 MB 4096x208
    float*    negm   = (float*)   (ws + 79183872);    //  16 KB

    k_prep   <<<NP,   256, 0, stream>>>(z1, z2, znp, zph, zpl);
    k_packW  <<<512,  256, 0, stream>>>(W1, W2, W1p, W2p);
    k_sim    <<<8192, 256, 0, stream>>>(zph, zpl, Spool);
    k_sort   <<<NP,   256, 0, stream>>>(Spool, sumexp, topIdx);
    k_mixproj<<<NP,   256, 0, stream>>>(z1, z2, b1, b2, W1p, W2p, znp,
                                        idx1, idx2, topIdx, negm);
    k_loss   <<<1,    256, 0, stream>>>(Spool, sumexp, negm, (float*)d_out);
}
